// GCN_67585605370170
// MI455X (gfx1250) — matrix operand order repaired and checked by reading, not compiled, not run on hardware
//
#include <hip/hip_runtime.h>
#include <stdint.h>

#define NN 8192   // nodes
#define NB 4      // batch
#define DD 64     // input feat
#define HH 128    // hidden 1
#define EE 256    // hidden 2
#define CC 16     // classes

typedef __attribute__((ext_vector_type(16))) __bf16 v16bf;
typedef __attribute__((ext_vector_type(8)))  float  v8f;
typedef __attribute__((ext_vector_type(4)))  unsigned int u32x4;

union BFrag { u32x4 q[2]; v16bf v; };

__device__ __forceinline__ unsigned short f32_to_bf16(float f) {
    union { float f; unsigned int u; } c; c.f = f;
    unsigned int u = c.u;
    u += 0x7FFFu + ((u >> 16) & 1u);   // round-to-nearest-even
    return (unsigned short)(u >> 16);
}

// ---------------------------------------------------------------- rowsum -> r
__global__ void k_rowsum(const float* __restrict__ adj, float* __restrict__ r) {
    __shared__ float red[256];
    int row = blockIdx.x;
    const float* p = adj + (size_t)row * NN;
    float s = 0.f;
    for (int j = threadIdx.x; j < NN; j += 256) s += p[j];
    red[threadIdx.x] = s; __syncthreads();
    for (int off = 128; off > 0; off >>= 1) {
        if ((int)threadIdx.x < off) red[threadIdx.x] += red[threadIdx.x + off];
        __syncthreads();
    }
    if (threadIdx.x == 0) {
        float rs = red[0];
        r[row] = (rs > 0.f) ? rsqrtf(rs) : 0.f;
    }
}

// --------------------------- adjn[i][j] = bf16(adj[j][i] * r[i] * r[j]) -----
__global__ void k_adjn(const float* __restrict__ adj, const float* __restrict__ r,
                       unsigned short* __restrict__ adjn) {
    __shared__ float tile[32][33];
    int i0 = blockIdx.x * 32;   // output rows (i)
    int j0 = blockIdx.y * 32;   // output cols (j)
    int tx = threadIdx.x, ty = threadIdx.y;   // 32 x 8
#pragma unroll
    for (int s = 0; s < 4; ++s) {
        int jl = ty + 8 * s;
        tile[jl][tx] = adj[(size_t)(j0 + jl) * NN + (i0 + tx)];
    }
    __syncthreads();
#pragma unroll
    for (int s = 0; s < 4; ++s) {
        int il = ty + 8 * s;
        float v = tile[tx][il] * r[i0 + il] * r[j0 + tx];
        adjn[(size_t)(i0 + il) * NN + (j0 + tx)] = f32_to_bf16(v);
    }
}

// ---------------------------------------------------------------- f32->bf16
__global__ void k_cvt_bf16(const float* __restrict__ in, unsigned short* __restrict__ out, int n) {
    int i = blockIdx.x * blockDim.x + threadIdx.x;
    if (i < n) out[i] = f32_to_bf16(in[i]);
}

// ------------- agg[b,i,d] = sum_j adjn[i,j]*xb[b,j,d] + xself[b,i,d] --------
// One wave: 16 nodes x 32 features x all 4 batches (A fragment reused 8x).
__global__ void __launch_bounds__(128)
k_agg_wmma(const unsigned short* __restrict__ adjn,
           const unsigned short* __restrict__ xb,     // [NB][NN][Dw] bf16
           const float*          __restrict__ xself,  // [NB][NN][Dw] f32
           float*                __restrict__ out,    // [NB][NN][Dw] f32
           int Dw) {
    const int wave = threadIdx.x >> 5;
    const int lane = threadIdx.x & 31;
    const int i0 = (blockIdx.x * 4 + wave) * 16;
    const int d0 = blockIdx.y * 32;

    v8f acc[NB][2];
#pragma unroll
    for (int b = 0; b < NB; ++b)
#pragma unroll
        for (int t = 0; t < 2; ++t)
#pragma unroll
            for (int e = 0; e < 8; ++e) acc[b][t][e] = 0.f;

    // A layout (16-bit 16x32): lane M = lane&15; K-halves depend on lane>=16
    const int m  = lane & 15;
    const int kb = (lane >> 4) * 8;
    const unsigned short* arow = adjn + (size_t)(i0 + m) * NN;

    for (int k = 0; k < NN; k += 32) {
        BFrag a;
        a.q[0] = *(const u32x4*)(arow + k + kb);        // K = kb..kb+7
        a.q[1] = *(const u32x4*)(arow + k + 16 + kb);   // K = 16+kb..
        __builtin_prefetch(arow + k + 512, 0, 1);       // stream-ahead hint
#pragma unroll
        for (int b = 0; b < NB; ++b) {
            // B layout (16-bit KxN): lane N = lane&15; K = kb+s, 16+kb+s (same as A)
            const unsigned short* brow = xb + ((size_t)b * NN + k + kb) * Dw + d0 + m;
#pragma unroll
            for (int t = 0; t < 2; ++t) {
                BFrag bf;
#pragma unroll
                for (int s = 0; s < 8; ++s) {
                    bf.v[s]     = __builtin_bit_cast(__bf16, brow[(size_t)s * Dw + t * 16]);
                    bf.v[8 + s] = __builtin_bit_cast(__bf16, brow[(size_t)(16 + s) * Dw + t * 16]);
                }
                acc[b][t] = __builtin_amdgcn_wmma_f32_16x16x32_bf16(
                    false, a.v, false, bf.v, (short)0, acc[b][t], false, false);
            }
        }
    }

    // C/D layout: lanes 0-15 -> N=lane, M=v; lanes 16-31 -> N=lane-16, M=v+8
    const int mhi  = (lane >> 4) * 8;
    const int dcol = lane & 15;
#pragma unroll
    for (int b = 0; b < NB; ++b)
#pragma unroll
        for (int t = 0; t < 2; ++t)
#pragma unroll
            for (int v = 0; v < 8; ++v) {
                int mm = v + mhi;
                int dd = d0 + t * 16 + dcol;
                size_t idx = ((size_t)b * NN + i0 + mm) * (size_t)Dw + dd;
                out[idx] = acc[b][t][v] + xself[idx];
            }
}

// ------- t = relu( l2norm_rows( agg @ W + bias ) ), one block per (b,n) -----
template <int DIN, int HOUT>
__global__ void __launch_bounds__(HOUT)
k_feat(const float* __restrict__ agg, const float* __restrict__ W,
       const float* __restrict__ bias, float* __restrict__ outp) {
    __shared__ float row[DIN];
    __shared__ float red[HOUT];
    int bn = blockIdx.x;           // b*NN + n
    int h  = threadIdx.x;          // 0..HOUT-1
    const float* ar = agg + (size_t)bn * DIN;
    if (h < DIN) row[h] = ar[h];
    __syncthreads();
    float acc = bias[h];
#pragma unroll 8
    for (int k = 0; k < DIN; ++k) acc += row[k] * W[k * HOUT + h];
    red[h] = acc * acc;
    __syncthreads();
    for (int off = HOUT / 2; off > 0; off >>= 1) {
        if (h < off) red[h] += red[h + off];
        __syncthreads();
    }
    float denom = fmaxf(sqrtf(red[0]), 1e-12f);
    outp[(size_t)bn * HOUT + h] = fmaxf(acc / denom, 0.f);
}

// ---- per-node BatchNorm over (batch, feature); writes f32 and opt. bf16 ----
template <int HOUT>
__global__ void k_bn(const float* __restrict__ t, float* __restrict__ gf,
                     unsigned short* __restrict__ gb) {
    __shared__ float rs[256], rq[256];
    int n = blockIdx.x, tid = threadIdx.x;
    float s = 0.f, q = 0.f;
    for (int i = tid; i < NB * HOUT; i += 256) {
        int b = i / HOUT, h = i - b * HOUT;
        float v = t[((size_t)b * NN + n) * HOUT + h];
        s += v; q += v * v;
    }
    rs[tid] = s; rq[tid] = q; __syncthreads();
    for (int off = 128; off > 0; off >>= 1) {
        if (tid < off) { rs[tid] += rs[tid + off]; rq[tid] += rq[tid + off]; }
        __syncthreads();
    }
    const float inv = 1.f / (float)(NB * HOUT);
    float mean = rs[0] * inv;
    float var  = rq[0] * inv - mean * mean;
    float sc   = rsqrtf(var + 1e-5f);
    for (int i = tid; i < NB * HOUT; i += 256) {
        int b = i / HOUT, h = i - b * HOUT;
        size_t idx = ((size_t)b * NN + n) * HOUT + h;
        float v = (t[idx] - mean) * sc;
        gf[idx] = v;
        if (gb) gb[idx] = f32_to_bf16(v);
    }
}

// ------------------------- pred = g @ Wp + bp (E=256 -> C=16) ---------------
__global__ void k_head(const float* __restrict__ g, const float* __restrict__ Wp,
                       const float* __restrict__ bp, float* __restrict__ pred) {
    int c  = threadIdx.x & 15;
    int nl = threadIdx.x >> 4;
    int bn = blockIdx.x * 16 + nl;         // over NB*NN
    const float* gr = g + (size_t)bn * EE;
    float acc = bp[c];
#pragma unroll 8
    for (int e = 0; e < EE; ++e) acc += gr[e] * Wp[e * CC + c];
    pred[(size_t)bn * CC + c] = acc;
}

// -------------------- softmax over node axis, in place on d_out -------------
__global__ void __launch_bounds__(1024) k_softmax(float* __restrict__ pred) {
    __shared__ float red[1024];
    int c = blockIdx.x, b = blockIdx.y, tid = threadIdx.x;
    float vals[8];
#pragma unroll
    for (int i = 0; i < 8; ++i)
        vals[i] = pred[((size_t)b * NN + tid + i * 1024) * CC + c];
    float mx = vals[0];
#pragma unroll
    for (int i = 1; i < 8; ++i) mx = fmaxf(mx, vals[i]);
    red[tid] = mx; __syncthreads();
    for (int off = 512; off > 0; off >>= 1) {
        if (tid < off) red[tid] = fmaxf(red[tid], red[tid + off]);
        __syncthreads();
    }
    mx = red[0]; __syncthreads();
    float s = 0.f;
#pragma unroll
    for (int i = 0; i < 8; ++i) { vals[i] = expf(vals[i] - mx); s += vals[i]; }
    red[tid] = s; __syncthreads();
    for (int off = 512; off > 0; off >>= 1) {
        if (tid < off) red[tid] += red[tid + off];
        __syncthreads();
    }
    float invs = 1.f / red[0];
#pragma unroll
    for (int i = 0; i < 8; ++i)
        pred[((size_t)b * NN + tid + i * 1024) * CC + c] = vals[i] * invs;
}

extern "C" void kernel_launch(void* const* d_in, const int* in_sizes, int n_in,
                              void* d_out, int out_size, void* d_ws, size_t ws_size,
                              hipStream_t stream) {
    const float* x   = (const float*)d_in[0];
    const float* adj = (const float*)d_in[1];
    const float* W1  = (const float*)d_in[2];
    const float* b1  = (const float*)d_in[3];
    const float* W2  = (const float*)d_in[4];
    const float* b2  = (const float*)d_in[5];
    const float* Wp  = (const float*)d_in[6];
    const float* bp  = (const float*)d_in[7];
    float* out = (float*)d_out;

    char* ws = (char*)d_ws;
    size_t off = 0;
    unsigned short* adjn = (unsigned short*)(ws + off); off += (size_t)NN * NN * 2;  // 128 MB
    float*          r    = (float*)(ws + off);          off += (size_t)NN * 4;
    unsigned short* xb   = (unsigned short*)(ws + off); off += (size_t)NB * NN * HH * 2; // 8 MB
    float*          agg  = (float*)(ws + off);          off += (size_t)NB * NN * HH * 4; // 16 MB
    float*          t    = (float*)(ws + off);          off += (size_t)NB * NN * EE * 4; // 32 MB
    float*          g    = (float*)(ws + off);          off += (size_t)NB * NN * EE * 4; // 32 MB

    // 1) r = rsqrt(rowsum), inf->0
    k_rowsum<<<NN, 256, 0, stream>>>(adj, r);
    // 2) adjn = bf16( adj^T * r_i * r_j )
    k_adjn<<<dim3(NN / 32, NN / 32), dim3(32, 8), 0, stream>>>(adj, r, adjn);
    // 3) x -> bf16 (B operand, layer 1)
    {
        int nx = NB * NN * DD;
        k_cvt_bf16<<<(nx + 255) / 256, 256, 0, stream>>>(x, xb, nx);
    }
    // 4) agg1 = adjn @ x + x   (WMMA bf16, f32 accumulate)
    k_agg_wmma<<<dim3(NN / 64, DD / 32), 128, 0, stream>>>(adjn, xb, x, agg, DD);
    // 5) t1 = relu(l2norm(agg1 @ W1 + b1))
    k_feat<DD, HH><<<NB * NN, HH, 0, stream>>>(agg, W1, b1, t);
    // 6) g1 = BN(t1)  (f32 into g, bf16 into xb for next WMMA)
    k_bn<HH><<<NN, 256, 0, stream>>>(t, g, xb);
    // 7) agg2 = adjn @ g1 + g1
    k_agg_wmma<<<dim3(NN / 64, HH / 32), 128, 0, stream>>>(adjn, xb, g, agg, HH);
    // 8) t2 = relu(l2norm(agg2 @ W2 + b2))
    k_feat<HH, EE><<<NB * NN, EE, 0, stream>>>(agg, W2, b2, t);
    // 9) g2 = BN(t2)
    k_bn<EE><<<NN, 256, 0, stream>>>(t, g, (unsigned short*)nullptr);
    // 10) pred = g2 @ Wp + bp -> d_out
    k_head<<<NB * NN / 16, 256, 0, stream>>>(g, Wp, bp, out);
    // 11) softmax over node axis, in place
    k_softmax<<<dim3(CC, NB), 1024, 0, stream>>>(out);
}
